// AutoSparseLinear_81243601371171
// MI455X (gfx1250) — matrix operand order repaired and checked by reading, not compiled, not run on hardware
//
#include <hip/hip_runtime.h>
#include <stdint.h>

// ---- problem sizes (fixed by the reference) ----
#define MSZ 4096
#define NSZ 4096
#define KSZ 4096
// ---- tiling ----
#define BM 128
#define BN 128
#define BK 32
#define BKP 40                  // padded bf16 stride: 80B/row -> 16B aligned, even bank spread
#define NKT (KSZ / BK)          // 128 K-steps

typedef __attribute__((ext_vector_type(16))) __bf16   v16bf;
typedef __attribute__((ext_vector_type(8)))  __bf16   v8bf;
typedef __attribute__((ext_vector_type(4)))  __bf16   v4bf;
typedef __attribute__((ext_vector_type(8)))  float    v8f;
typedef __attribute__((ext_vector_type(4)))  float    f32x4;
typedef __attribute__((ext_vector_type(4)))  uint32_t u32x4;

__device__ __forceinline__ float bf16_bits_to_f32(unsigned short h) {
  return __builtin_bit_cast(float, (uint32_t)h << 16);
}
__device__ __forceinline__ __bf16 bf16_from_bits(unsigned short h) {
  return __builtin_bit_cast(__bf16, h);
}
// hi = truncate(f), lo = truncate(f - hi): 3 VALU (mov_b16 / fma_mix / mov_b16).
__device__ __forceinline__ void split_bf16(float f, __bf16& hi, __bf16& lo) {
  uint32_t u = __builtin_bit_cast(uint32_t, f);
  unsigned short hb = (unsigned short)(u >> 16);
  float rem = f - bf16_bits_to_f32(hb);
  unsigned short lb = (unsigned short)(__builtin_bit_cast(uint32_t, rem) >> 16);
  hi = bf16_from_bits(hb);
  lo = bf16_from_bits(lb);
}

__device__ __forceinline__ v16bf ld_frag(const __bf16* p) {
  v8bf a = *(const v8bf*)(p);        // 16B aligned
  v8bf b = *(const v8bf*)(p + 8);
  return __builtin_shufflevector(a, b, 0,1,2,3,4,5,6,7,8,9,10,11,12,13,14,15);
}

// ---------------- prep: W[K][N] f32 -> Wt_hi/Wt_lo[N][K] bf16 (one-time) ----------------
__global__ __launch_bounds__(256)
void wt_prep(const float* __restrict__ W, __bf16* __restrict__ Wth, __bf16* __restrict__ Wtl) {
  __shared__ float Ts[64][68];
  const int tid = threadIdx.x;
  const int n0 = blockIdx.x * 64;
  const int k0 = blockIdx.y * 64;
  const int c  = (tid & 15) * 4;
  const int r  = tid >> 4;                  // 0..15
#pragma unroll
  for (int p = 0; p < 4; ++p)
    *(f32x4*)&Ts[r + p * 16][c] = *(const f32x4*)(W + (size_t)(k0 + r + p * 16) * NSZ + n0 + c);
  __syncthreads();
#pragma unroll
  for (int p = 0; p < 4; ++p) {
    const int nl = r + p * 16;
    v4bf hv, lv;
#pragma unroll
    for (int j = 0; j < 4; ++j) {
      __bf16 h, l;
      split_bf16(Ts[c + j][nl], h, l);
      hv[j] = h; lv[j] = l;
    }
    *(v4bf*)(Wth + (size_t)(n0 + nl) * KSZ + k0 + c) = hv;
    *(v4bf*)(Wtl + (size_t)(n0 + nl) * KSZ + k0 + c) = lv;
  }
}

// ---------------- main GEMM ----------------
// USE_WT=true : B tiles are a pure bf16 copy from pre-decomposed Wt (d_ws path).
// USE_WT=false: B tiles converted + transposed in-LDS from f32 W (no-workspace fallback).
template <bool USE_WT>
__global__ __launch_bounds__(256)
void sparse_linear_wmma(const float* __restrict__ A, const float* __restrict__ Wf,
                        const __bf16* __restrict__ Wth, const __bf16* __restrict__ Wtl,
                        const float* __restrict__ bias, float* __restrict__ out) {
  // A stored in contiguous k order: Ah[m][khalf*16 + e] (e = operand element)
  __shared__ __bf16 Ah[BM][BKP];
  __shared__ __bf16 Al[BM][BKP];
  // B stored K-major: Bh[n][k] (operand element e == k for lane khalf -> contiguous)
  __shared__ __bf16 Bh[BN][BKP];
  __shared__ __bf16 Bl[BN][BKP];

  const int tid   = threadIdx.x;
  const int lane  = tid & 31;
  const int wave  = tid >> 5;
  const int wr    = wave >> 1;     // 0..3
  const int wc    = wave & 1;      // 0..1
  const int laneM = lane & 15;
  const int khalf = lane >> 4;

  const int rowBase = blockIdx.y * BM;
  const int colBase = blockIdx.x * BN;

  // A staging pattern: 8 threads per 32-float row, 4 row-passes
  const int arow = tid >> 3;
  const int ak   = (tid & 7) * 4;
  const int asid = ak; // store slot (contiguous k)

  // B staging pattern (USE_WT): 2 threads per 128-row, 2 b128 chunks each
  const int bn  = tid >> 1;
  const int bc0 = (tid & 1) * 2;
  // B staging pattern (!USE_WT): 32 threads per 128-float W row, 4 k-passes
  const int bkr = tid >> 5;
  const int bnc = (tid & 31) * 4;

  v8f acc[2][4];
#pragma unroll
  for (int mi = 0; mi < 2; ++mi)
#pragma unroll
    for (int ni = 0; ni < 4; ++ni)
#pragma unroll
      for (int e = 0; e < 8; ++e) acc[mi][ni][e] = 0.0f;

  f32x4 pa[4];
  u32x4 pbh[2], pbl[2];
  f32x4 pbf[4];

  // ---- prefetch tile 0 into registers ----
#pragma unroll
  for (int p = 0; p < 4; ++p)
    pa[p] = *(const f32x4*)(A + (size_t)(rowBase + arow + p * 32) * KSZ + ak);
  if (USE_WT) {
#pragma unroll
    for (int q = 0; q < 2; ++q) {
      pbh[q] = *(const u32x4*)(Wth + (size_t)(colBase + bn) * KSZ + (bc0 + q) * 8);
      pbl[q] = *(const u32x4*)(Wtl + (size_t)(colBase + bn) * KSZ + (bc0 + q) * 8);
    }
  } else {
#pragma unroll
    for (int p = 0; p < 4; ++p)
      pbf[p] = *(const f32x4*)(Wf + (size_t)(bkr + p * 8) * NSZ + colBase + bnc);
  }

  for (int kt = 0; kt < NKT; ++kt) {
    // ---- stage current tile into LDS ----
#pragma unroll
    for (int p = 0; p < 4; ++p) {        // A: split f32 -> hi/lo bf16, contiguous k slot
      v4bf hv, lv;
#pragma unroll
      for (int j = 0; j < 4; ++j) {
        __bf16 h, l;
        split_bf16(pa[p][j], h, l);
        hv[j] = h; lv[j] = l;
      }
      *(v4bf*)&Ah[arow + p * 32][asid] = hv;
      *(v4bf*)&Al[arow + p * 32][asid] = lv;
    }
    if (USE_WT) {
#pragma unroll
      for (int q = 0; q < 2; ++q) {      // B: raw bf16 b128 copies
        *(u32x4*)&Bh[bn][(bc0 + q) * 8] = pbh[q];
        *(u32x4*)&Bl[bn][(bc0 + q) * 8] = pbl[q];
      }
    } else {
#pragma unroll
      for (int p = 0; p < 4; ++p) {      // B: convert + transpose scatter
#pragma unroll
        for (int i = 0; i < 4; ++i) {
          __bf16 h, l;
          split_bf16(pbf[p][i], h, l);
          Bh[bnc + i][bkr + p * 8] = h;
          Bl[bnc + i][bkr + p * 8] = l;
        }
      }
    }
    __syncthreads();

    // ---- issue global loads for next tile (overlap with WMMAs) ----
    if (kt + 1 < NKT) {
      const int k1 = (kt + 1) * BK;
#pragma unroll
      for (int p = 0; p < 4; ++p)
        pa[p] = *(const f32x4*)(A + (size_t)(rowBase + arow + p * 32) * KSZ + k1 + ak);
      if (USE_WT) {
#pragma unroll
        for (int q = 0; q < 2; ++q) {
          pbh[q] = *(const u32x4*)(Wth + (size_t)(colBase + bn) * KSZ + k1 + (bc0 + q) * 8);
          pbl[q] = *(const u32x4*)(Wtl + (size_t)(colBase + bn) * KSZ + k1 + (bc0 + q) * 8);
        }
      } else {
#pragma unroll
        for (int p = 0; p < 4; ++p)
          pbf[p] = *(const f32x4*)(Wf + (size_t)(k1 + bkr + p * 8) * NSZ + colBase + bnc);
      }
    }
    if (kt + 2 < NKT) {                  // gfx1250 global_prefetch two tiles ahead
      const int k2 = (kt + 2) * BK;
      __builtin_prefetch(A + (size_t)(rowBase + arow) * KSZ + k2 + ak, 0, 3);
      if (USE_WT)
        __builtin_prefetch(Wth + (size_t)(colBase + bn) * KSZ + k2, 0, 3);
      else
        __builtin_prefetch(Wf + (size_t)(k2 + bkr) * NSZ + colBase + bnc, 0, 3);
    }

    // ---- fragments (already bf16 in operand order) + 3-term split WMMA ----
    v16bf ahf[2], alf[2];
#pragma unroll
    for (int mi = 0; mi < 2; ++mi) {
      const int row = wr * 32 + mi * 16 + laneM;
      ahf[mi] = ld_frag(&Ah[row][khalf * 16]);
      alf[mi] = ld_frag(&Al[row][khalf * 16]);
    }
#pragma unroll
    for (int ni = 0; ni < 4; ++ni) {
      const int col = wc * 64 + ni * 16 + laneM;
      v16bf bhf = ld_frag(&Bh[col][khalf * 16]);
      v16bf blf = ld_frag(&Bl[col][khalf * 16]);
#pragma unroll
      for (int mi = 0; mi < 2; ++mi) {
        acc[mi][ni] = __builtin_amdgcn_wmma_f32_16x16x32_bf16(
            false, ahf[mi], false, bhf, (short)0, acc[mi][ni], false, false);
        acc[mi][ni] = __builtin_amdgcn_wmma_f32_16x16x32_bf16(
            false, ahf[mi], false, blf, (short)0, acc[mi][ni], false, false);
        acc[mi][ni] = __builtin_amdgcn_wmma_f32_16x16x32_bf16(
            false, alf[mi], false, bhf, (short)0, acc[mi][ni], false, false);
      }
    }
    __syncthreads();
  }

  // ---- epilogue: D layout is VGPR r -> row (r + 8*khalf), lane -> col ----
#pragma unroll
  for (int mi = 0; mi < 2; ++mi) {
#pragma unroll
    for (int ni = 0; ni < 4; ++ni) {
      const int gcol = colBase + wc * 64 + ni * 16 + laneM;
      const int grow = rowBase + wr * 32 + mi * 16 + 8 * khalf;
      const float bv = bias[gcol];
      float* op = out + (size_t)grow * NSZ + gcol;
#pragma unroll
      for (int r = 0; r < 8; ++r)
        op[(size_t)r * NSZ] = acc[mi][ni][r] + bv;
    }
  }
}

extern "C" void kernel_launch(void* const* d_in, const int* in_sizes, int n_in,
                              void* d_out, int out_size, void* d_ws, size_t ws_size,
                              hipStream_t stream) {
  (void)in_sizes; (void)n_in; (void)out_size;
  const float* A    = (const float*)d_in[0];  // sparse activations [4096,4096]
  const float* W    = (const float*)d_in[1];  // weight [4096,4096] (K-major)
  const float* bias = (const float*)d_in[2];  // [4096]
  float* out = (float*)d_out;

  dim3 grid(NSZ / BN, MSZ / BM);              // 32 x 32 blocks, 256 threads (8 waves)
  const size_t need = (size_t)KSZ * NSZ * sizeof(__bf16) * 2;  // Wt_hi + Wt_lo = 64MB

  if (ws_size >= need) {
    __bf16* Wth = (__bf16*)d_ws;
    __bf16* Wtl = Wth + (size_t)KSZ * NSZ;
    wt_prep<<<dim3(NSZ / 64, KSZ / 64), 256, 0, stream>>>(W, Wth, Wtl);
    sparse_linear_wmma<true><<<grid, 256, 0, stream>>>(A, nullptr, Wth, Wtl, bias, out);
  } else {
    sparse_linear_wmma<false><<<grid, 256, 0, stream>>>(A, W, nullptr, nullptr, bias, out);
  }
}
